// pretrainModel_80049600463489
// MI455X (gfx1250) — matrix operand order repaired and checked by reading, not compiled, not run on hardware
//
#include <hip/hip_runtime.h>
#include <math.h>

// ============================================================================
// Perceiver diffusion forward pass for MI455X (gfx1250, wave32, WMMA bf16,
// Tensor Data Mover staging).
//
// Input order assumption: top-level setup_inputs() dict order, nested params
// flattened jax-pytree style (sorted keys, recursively).
// ============================================================================

typedef unsigned short u16;
typedef unsigned int u32;
typedef unsigned long long u64;
typedef __bf16 bf16x16 __attribute__((ext_vector_type(16)));
typedef float f32x8 __attribute__((ext_vector_type(8)));
typedef unsigned int u32x4 __attribute__((ext_vector_type(4)));
typedef int i32x4 __attribute__((ext_vector_type(4)));
typedef int i32x8 __attribute__((ext_vector_type(8)));

union Frag { uint4 q[2]; bf16x16 v; };

__device__ __forceinline__ u16 f2bf(float x) {
    u32 u = __builtin_bit_cast(u32, x);
    u32 r = u + 0x7FFFu + ((u >> 16) & 1u);   // round-to-nearest-even
    return (u16)(r >> 16);
}
__device__ __forceinline__ float bf2f(u16 h) {
    return __builtin_bit_cast(float, (u32)h << 16);
}
__device__ __forceinline__ float mishf(float x) {
    float sp = (x > 20.f) ? x : log1pf(__expf(x));
    return x * tanhf(sp);
}
__device__ __forceinline__ float gelu_tanh(float x) {
    return 0.5f * x * (1.f + tanhf(0.7978845608f * (x + 0.044715f * x * x * x)));
}
__device__ __forceinline__ f32x8 zero8() {
    f32x8 v = {0.f,0.f,0.f,0.f,0.f,0.f,0.f,0.f};
    return v;
}
__device__ __forceinline__ f32x8 wmma_bf16(const Frag& a, const Frag& b, f32x8 c) {
    return __builtin_amdgcn_wmma_f32_16x16x32_bf16(false, a.v, false, b.v,
                                                   (short)0, c, false, false);
}

// ---------------------------------------------------------------------------
// Tensor Data Mover: async 2D tile (bf16 elements) global -> LDS.
//   lds_off     : LDS byte offset of destination (tile stored packed,
//                 tile_d0 elements per row, rows consecutive)
//   gaddr       : global byte address of tile start
//   tile_d0/d1  : tile extent (elements / rows); d0*2 bytes, mult of 4
//   tens_d0/d1  : remaining tensor extent from tile start (OOB reads -> 0)
//   stride0     : elements between consecutive rows in memory
// D# layout per CDNA5 ISA 8.3/8.4 (2D: groups 2,3 zero).
// This toolchain exposes the 6-arg builtin:
//   (uint32x4 g0, int32x8 g1, int32x4 g2, int32x4 g3, int32x8 pad, i32 cpol)
// ---------------------------------------------------------------------------
__device__ __forceinline__ void tdm_load_2d(u32 lds_off, const void* gaddr,
                                            u32 tile_d0, u32 tile_d1,
                                            u32 tens_d0, u32 tens_d1,
                                            u64 stride0) {
    u64 ga = (u64)(size_t)gaddr;
    u32x4 g0;
    g0[0] = 1u;                                        // count=1, user D#
    g0[1] = lds_off;                                   // lds_addr
    g0[2] = (u32)(ga & 0xFFFFFFFFu);                   // global_addr[31:0]
    g0[3] = (u32)((ga >> 32) & 0x01FFFFFFu) | (2u << 30);  // addr[56:32]|type=2
    i32x8 g1;
    g1[0] = (int)(1u << 16);                           // data_size=1 (2 bytes)
    g1[1] = (int)((tens_d0 & 0xFFFFu) << 16);          // tensor_dim0[15:0]
    g1[2] = (int)(((tens_d0 >> 16) & 0xFFFFu) | ((tens_d1 & 0xFFFFu) << 16));
    g1[3] = (int)(((tens_d1 >> 16) & 0xFFFFu) | ((tile_d0 & 0xFFFFu) << 16));
    g1[4] = (int)(tile_d1 & 0xFFFFu);                  // tile_dim1 (tile_dim2=0)
    g1[5] = (int)(u32)(stride0 & 0xFFFFFFFFu);         // tensor_dim0_stride lo
    g1[6] = (int)(u32)((stride0 >> 32) & 0xFFFFu);     // stride hi (dim1_str=0)
    g1[7] = 0;
    i32x4 z4 = {0, 0, 0, 0};
    i32x8 z8 = {0, 0, 0, 0, 0, 0, 0, 0};
    __builtin_amdgcn_tensor_load_to_lds(g0, g1, z4, z4, z8, 0);
}
__device__ __forceinline__ u32 lds_offset_of(const void* p) {
    return (u32)(size_t)p;   // LDS aperture: addr[31:0] is the LDS byte offset
}

// A-fragment (16x32 bf16), CDNA5 16-bit A layout.
__device__ __forceinline__ Frag load_a_frag(const u16* base, int rowStride, int kb) {
    int lane = threadIdx.x & 31;
    int m = lane & 15, kh = lane >> 4;
    Frag f;
    f.q[0] = *(const uint4*)(base + (size_t)m * rowStride + kb + 8 * kh);
    f.q[1] = *(const uint4*)(base + (size_t)m * rowStride + kb + 16 + 8 * kh);
    return f;
}
// B-fragment (32x16 bf16) from a K-contiguous ("Bt[n][k]") tile.
__device__ __forceinline__ Frag load_b_frag(const u16* base, int rowStride, int kb) {
    int lane = threadIdx.x & 31;
    int n = lane & 15, kh = lane >> 4;
    const u16* p = base + (size_t)n * rowStride + kb + 8 * kh;
    Frag f;
    f.q[0] = *(const uint4*)p;
    f.q[1] = *(const uint4*)(p + 16);
    return f;
}

__device__ __forceinline__ void blockReduceSum2(float& a, float& b) {
    #pragma unroll
    for (int off = 16; off; off >>= 1) {
        a += __shfl_xor(a, off, 32);
        b += __shfl_xor(b, off, 32);
    }
    __shared__ float ra[8], rb[8];
    int w = threadIdx.x >> 5, lane = threadIdx.x & 31;
    if (lane == 0) { ra[w] = a; rb[w] = b; }
    __syncthreads();
    a = 0.f; b = 0.f;
    #pragma unroll
    for (int i = 0; i < 8; i++) { a += ra[i]; b += rb[i]; }
}

// ---------------------------------------------------------------------------
// float -> bf16 conversion (plain, and transposing [K,N] -> [N,K])
// ---------------------------------------------------------------------------
__global__ __launch_bounds__(256) void cvt_kernel(const float* __restrict__ in,
                                                  u16* __restrict__ out, int n) {
    int i = blockIdx.x * 256 + threadIdx.x;
    if (i < n) out[i] = f2bf(in[i]);
}
__global__ __launch_bounds__(256) void cvt_t_kernel(const float* __restrict__ in,
                                                    u16* __restrict__ out,
                                                    int K, int N) {
    int i = blockIdx.x * 256 + threadIdx.x;
    if (i >= K * N) return;
    int n = i / K, k = i - n * K;
    out[i] = f2bf(in[(size_t)k * N + n]);
}

// ---------------------------------------------------------------------------
// codebook gather: out[row, d] = bf16(codebook[idx[row], d])
// ---------------------------------------------------------------------------
__global__ __launch_bounds__(256) void gather_kernel(const int* __restrict__ idx,
                                                     const float* __restrict__ cb,
                                                     u16* __restrict__ out) {
    int row = blockIdx.x, d = threadIdx.x;
    int c = idx[row];
    out[(size_t)row * 256 + d] = f2bf(cb[(size_t)c * 256 + d]);
}

// ---------------------------------------------------------------------------
// time-embedding + prompt-embedding MLPs (tiny; one block per batch)
// ---------------------------------------------------------------------------
__global__ __launch_bounds__(256) void embed_small_kernel(
    const float* __restrict__ timev, const float* __restrict__ cond,
    const float* __restrict__ tw1, const float* __restrict__ tb1,
    const float* __restrict__ tw2, const float* __restrict__ tb2,
    const float* __restrict__ plng, const float* __restrict__ plnb,
    const float* __restrict__ pw1, const float* __restrict__ pb1,
    const float* __restrict__ pw2, const float* __restrict__ pb2,
    float* __restrict__ te, float* __restrict__ pe) {
    int b = blockIdx.x, tid = threadIdx.x;
    __shared__ float s[256];
    __shared__ float h1[1024];
    __shared__ float ch[1024];
    __shared__ float h2[256];
    __shared__ float rs[8], rq[8];

    if (tid < 128) {
        float f = __expf(-logf(10000.f) * (float)tid / 127.f);
        float e = timev[b] * f;
        s[tid] = __sinf(e);
        s[tid + 128] = __cosf(e);
    }
    __syncthreads();
    for (int j = tid; j < 1024; j += 256) {
        float acc = tb1[j];
        for (int d = 0; d < 256; ++d) acc += s[d] * tw1[d * 1024 + j];
        h1[j] = mishf(acc);
    }
    float ls = 0.f, lq = 0.f;
    for (int j = tid; j < 1024; j += 256) {
        float c = cond[b * 1024 + j];
        ls += c; lq += c * c;
    }
    #pragma unroll
    for (int off = 16; off; off >>= 1) {
        ls += __shfl_xor(ls, off, 32);
        lq += __shfl_xor(lq, off, 32);
    }
    int w = tid >> 5, lane = tid & 31;
    if (lane == 0) { rs[w] = ls; rq[w] = lq; }
    __syncthreads();
    ls = 0.f; lq = 0.f;
    #pragma unroll
    for (int i = 0; i < 8; i++) { ls += rs[i]; lq += rq[i]; }
    float mean = ls * (1.f / 1024.f);
    float var = lq * (1.f / 1024.f) - mean * mean;
    float inv = rsqrtf(var + 1e-5f);
    for (int j = tid; j < 1024; j += 256) {
        float c = cond[b * 1024 + j];
        ch[j] = (c - mean) * inv * plng[j] + plnb[j];
    }
    __syncthreads();
    {
        float acc = tb2[tid];
        for (int j = 0; j < 1024; ++j) acc += h1[j] * tw2[j * 256 + tid];
        te[b * 256 + tid] = acc;
    }
    {
        float acc = pb1[tid];
        for (int j = 0; j < 1024; ++j) acc += ch[j] * pw1[j * 256 + tid];
        h2[tid] = mishf(acc);
    }
    __syncthreads();
    {
        float acc = pb2[tid];
        for (int j = 0; j < 256; ++j) acc += h2[j] * pw2[j * 256 + tid];
        pe[b * 256 + tid] = acc;
    }
}

// ---------------------------------------------------------------------------
// stacked = LN_eln( te * (pe * base) + pos ), base = {te, pe, tre[s-2]}
// ---------------------------------------------------------------------------
__global__ __launch_bounds__(256) void build_stack_kernel(
    float* __restrict__ stacked, const float* __restrict__ te,
    const float* __restrict__ pe, const float* __restrict__ pos,
    const float* __restrict__ g, const float* __restrict__ beta) {
    int row = blockIdx.x;
    int b = row / 2306, s = row % 2306, d = threadIdx.x;
    float base;
    if (s == 0) base = te[b * 256 + d];
    else if (s == 1) base = pe[b * 256 + d];
    else base = stacked[(size_t)row * 256 + d];
    float v = pe[b * 256 + d] * base;
    v = te[b * 256 + d] * v + pos[s * 256 + d];
    float a = v, q = v * v;
    blockReduceSum2(a, q);
    float mean = a * (1.f / 256.f);
    float var = q * (1.f / 256.f) - mean * mean;
    float y = (v - mean) * rsqrtf(var + 1e-5f) * g[d] + beta[d];
    stacked[(size_t)row * 256 + d] = y;
}

// ---------------------------------------------------------------------------
// generic row-LayerNorm (DIM=256): f32 in -> bf16 out
// ---------------------------------------------------------------------------
__global__ __launch_bounds__(256) void ln_kernel(const float* __restrict__ in,
                                                 u16* __restrict__ out,
                                                 const float* __restrict__ g,
                                                 const float* __restrict__ beta) {
    size_t row = blockIdx.x;
    int d = threadIdx.x;
    float x = in[row * 256 + d];
    float a = x, q = x * x;
    blockReduceSum2(a, q);
    float mean = a * (1.f / 256.f);
    float var = q * (1.f / 256.f) - mean * mean;
    float y = (x - mean) * rsqrtf(var + 1e-5f) * g[d] + beta[d];
    out[row * 256 + d] = f2bf(y);
}

// ---------------------------------------------------------------------------
// GEGLU
// ---------------------------------------------------------------------------
__global__ __launch_bounds__(256) void geglu_kernel(const u16* __restrict__ u,
                                                    u16* __restrict__ h,
                                                    int Mrows) {
    size_t i = (size_t)blockIdx.x * 256 + threadIdx.x;
    size_t total = (size_t)Mrows * 1024;
    if (i >= total) return;
    size_t m = i >> 10;
    int n = (int)(i & 1023);
    float a = bf2f(u[m * 2048 + n]);
    float g = bf2f(u[m * 2048 + 1024 + n]);
    h[i] = f2bf(a * gelu_tanh(g));
}

// ---------------------------------------------------------------------------
// WMMA bf16 GEMM, TDM-staged, double-buffered.
//   A : [*,K] bf16 row-major, W : [N,K] bf16 (PRE-TRANSPOSED, K-contiguous)
// Tiles never straddle batches (all rows-per-batch are multiples of 128 or
// grid-exact), so the batch row-remap is affine per block.
// ---------------------------------------------------------------------------
__global__ __launch_bounds__(256) void gemm_bf16_kernel(
    const u16* __restrict__ A, const u16* __restrict__ Wt,
    const float* __restrict__ bias, const float* __restrict__ residual,
    float* __restrict__ outF, u16* __restrict__ outH,
    int M, int N, int Kd, int aPhysRows,
    int ropb, int ripb, int rskip, int oripb, int oskip) {
    __shared__ __align__(16) u16 As[2][128 * 32];
    __shared__ __align__(16) u16 Bs[2][128 * 32];
    int tid = threadIdx.x;
    int w = tid >> 5;
    int wm = w & 3, wn = w >> 2;
    int m0 = blockIdx.y * 128, n0 = blockIdx.x * 128;

    int bb = m0 / ropb, br = m0 - bb * ropb;        // per-block affine remap
    int pr0 = bb * ripb + rskip + br;               // physical A row of tile
    int or0 = bb * oripb + oskip + br;              // physical out row of tile
    u32 rowsRemA = (u32)(aPhysRows - pr0);
    u32 rowsRemB = (u32)(N - n0);

    f32x8 acc[2][4];
    #pragma unroll
    for (int mi = 0; mi < 2; ++mi)
        #pragma unroll
        for (int ni = 0; ni < 4; ++ni) acc[mi][ni] = zero8();

    const int nk = Kd >> 5;
    if (tid < 32) {   // wave 0 drives the Tensor Data Mover
        tdm_load_2d(lds_offset_of(As[0]), A + (size_t)pr0 * Kd, 32, 128,
                    (u32)Kd, rowsRemA, (u64)Kd);
        tdm_load_2d(lds_offset_of(Bs[0]), Wt + (size_t)n0 * Kd, 32, 128,
                    (u32)Kd, rowsRemB, (u64)Kd);
    }
    for (int ik = 0; ik < nk; ++ik) {
        int buf = ik & 1;
        int k0n = (ik + 1) << 5;
        if (tid < 32) {
            if (ik + 1 < nk) {   // prefetch next tile into the other buffer
                tdm_load_2d(lds_offset_of(As[buf ^ 1]),
                            A + (size_t)pr0 * Kd + k0n, 32, 128,
                            (u32)(Kd - k0n), rowsRemA, (u64)Kd);
                tdm_load_2d(lds_offset_of(Bs[buf ^ 1]),
                            Wt + (size_t)n0 * Kd + k0n, 32, 128,
                            (u32)(Kd - k0n), rowsRemB, (u64)Kd);
                __builtin_amdgcn_s_wait_tensorcnt(2);  // current tile done
            } else {
                __builtin_amdgcn_s_wait_tensorcnt(0);
            }
        }
        __syncthreads();   // tile `buf` ready for all waves
        Frag af[2], bfr[4];
        #pragma unroll
        for (int mi = 0; mi < 2; ++mi)
            af[mi] = load_a_frag(As[buf] + (wm * 32 + mi * 16) * 32, 32, 0);
        #pragma unroll
        for (int ni = 0; ni < 4; ++ni)
            bfr[ni] = load_b_frag(Bs[buf] + (wn * 64 + ni * 16) * 32, 32, 0);
        #pragma unroll
        for (int mi = 0; mi < 2; ++mi)
            #pragma unroll
            for (int ni = 0; ni < 4; ++ni)
                acc[mi][ni] = wmma_bf16(af[mi], bfr[ni], acc[mi][ni]);
        __syncthreads();   // done reading `buf` before it is re-targeted
    }

    int lane = tid & 31;
    int lr8 = (lane >> 4) * 8;
    int lc = lane & 15;
    #pragma unroll
    for (int mi = 0; mi < 2; ++mi) {
        #pragma unroll
        for (int ni = 0; ni < 4; ++ni) {
            int cb = n0 + wn * 64 + ni * 16 + lc;
            float bv = (bias != nullptr && cb < N) ? bias[cb] : 0.f;
            #pragma unroll
            for (int r = 0; r < 8; ++r) {
                int rr = m0 + wm * 32 + mi * 16 + lr8 + r;
                if (rr < M && cb < N) {
                    size_t oi = (size_t)(or0 + (rr - m0)) * (size_t)N + cb;
                    float v = acc[mi][ni][r] + bv;
                    if (residual) v += residual[oi];
                    if (outF) outF[oi] = v;
                    else outH[oi] = f2bf(v);
                }
            }
        }
    }
}

// ---------------------------------------------------------------------------
// Fused flash-attention (single head, DH in {32,64}), bf16 in/out.
// 8 waves/block, 16 queries/wave, ctx streamed in chunks of 32.
// K chunk staged by TDM (OOB tail auto-zeroed); V transposed manually.
// ---------------------------------------------------------------------------
template <int DH>
__global__ __launch_bounds__(256) void attn_kernel(
    const u16* __restrict__ Q, const u16* __restrict__ Kt,
    const u16* __restrict__ Vt, u16* __restrict__ O,
    int NQ, int NC, float scale) {
    constexpr int NF = DH / 32;
    constexpr int NO = DH / 16;
    __shared__ __align__(16) u16 Ks[32 * DH];       // [ctx 32][dh] row-major
    __shared__ __align__(16) u16 Vs[DH * 32];       // [dh][ctx] K-contiguous
    __shared__ __align__(16) u16 Ps[8 * 16 * 32];   // per-wave P tiles

    int b = blockIdx.y;
    int tid = threadIdx.x, w = tid >> 5, lane = tid & 31;
    int q0 = (blockIdx.x * 8 + w) * 16;
    if (q0 > NQ - 16) q0 = NQ - 16;

    const u16* Qb = Q + ((size_t)b * NQ + q0) * DH;
    const u16* Kb = Kt + (size_t)b * NC * DH;
    const u16* Vb = Vt + (size_t)b * NC * DH;

    Frag aQ[NF];
    #pragma unroll
    for (int f = 0; f < NF; ++f) aQ[f] = load_a_frag(Qb, DH, f * 32);

    float mrun[8], lrun[8];
    #pragma unroll
    for (int r = 0; r < 8; ++r) { mrun[r] = -1e30f; lrun[r] = 0.f; }
    f32x8 oacc[NO];
    #pragma unroll
    for (int ni = 0; ni < NO; ++ni) oacc[ni] = zero8();

    u16* Pw = Ps + w * 16 * 32;
    int lhi = lane >> 4, llo = lane & 15;

    for (int c0 = 0; c0 < NC; c0 += 32) {
        __syncthreads();   // previous chunk fully consumed
        if (tid < 32)      // TDM: K chunk, rows beyond NC read as zero
            tdm_load_2d(lds_offset_of(Ks), Kb + (size_t)c0 * DH, DH, 32,
                        (u32)DH, (u32)(NC - c0), (u64)DH);
        for (int i = tid; i < 32 * DH; i += 256) {   // V chunk, transposed
            int kk = i / DH, d = i % DH;
            u16 v = 0;
            if (c0 + kk < NC) v = Vb[(size_t)(c0 + kk) * DH + d];
            Vs[d * 32 + kk] = v;
        }
        if (tid < 32) __builtin_amdgcn_s_wait_tensorcnt(0);
        __syncthreads();

        f32x8 s0 = zero8(), s1 = zero8();
        #pragma unroll
        for (int f = 0; f < NF; ++f) {
            s0 = wmma_bf16(aQ[f], load_b_frag(Ks, DH, f * 32), s0);
            s1 = wmma_bf16(aQ[f], load_b_frag(Ks + 16 * DH, DH, f * 32), s1);
        }

        #pragma unroll
        for (int r = 0; r < 8; ++r) {
            float v0 = s0[r] * scale, v1 = s1[r] * scale;
            v0 = (c0 + llo < NC) ? v0 : -1e30f;
            v1 = (c0 + 16 + llo < NC) ? v1 : -1e30f;
            float t = fmaxf(v0, v1);
            #pragma unroll
            for (int off = 1; off < 16; off <<= 1) t = fmaxf(t, __shfl_xor(t, off, 32));
            float mn = fmaxf(mrun[r], t);
            float alpha = __expf(mrun[r] - mn);
            float p0 = __expf(v0 - mn), p1 = __expf(v1 - mn);
            float rsm = p0 + p1;
            #pragma unroll
            for (int off = 1; off < 16; off <<= 1) rsm += __shfl_xor(rsm, off, 32);
            lrun[r] = lrun[r] * alpha + rsm;
            mrun[r] = mn;
            #pragma unroll
            for (int ni = 0; ni < NO; ++ni) oacc[ni][r] *= alpha;
            int ml = r + (lhi << 3);
            Pw[ml * 32 + llo] = f2bf(p0);
            Pw[ml * 32 + 16 + llo] = f2bf(p1);
        }

        Frag aP = load_a_frag(Pw, 32, 0);
        #pragma unroll
        for (int ni = 0; ni < NO; ++ni)
            oacc[ni] = wmma_bf16(aP, load_b_frag(Vs + (ni * 16) * 32, 32, 0), oacc[ni]);
    }

    #pragma unroll
    for (int r = 0; r < 8; ++r) {
        float inv = (lrun[r] > 0.f) ? 1.f / lrun[r] : 0.f;
        int row = q0 + r + (lhi << 3);
        #pragma unroll
        for (int ni = 0; ni < NO; ++ni) {
            O[((size_t)b * NQ + row) * DH + ni * 16 + llo] = f2bf(oacc[ni][r] * inv);
        }
    }
}

// ===========================================================================
// Host orchestration
// ===========================================================================
enum {
    I_TRAJ = 0, I_TIME, I_COND, I_VALUE, I_XCOND, I_CODEBOOK,
    I_CA_BO, I_CA_CLNB, I_CA_CLNG, I_CA_LNB, I_CA_LNG,
    I_CA_WK, I_CA_WO, I_CA_WQ, I_CA_WV,
    I_CF_B1, I_CF_B2, I_CF_LNB, I_CF_LNG, I_CF_W1, I_CF_W2,
    I_DC_BO, I_DC_CLNB, I_DC_CLNG, I_DC_LNB, I_DC_LNG,
    I_DC_WK, I_DC_WO, I_DC_WQ, I_DC_WV,
    I_ELN_B, I_ELN_G,
    I_P_B1, I_P_B2, I_P_LNB, I_P_LNG, I_P_W1, I_P_W2,
    I_POS, I_PR_B, I_PR_W,
    I_SA_BO, I_SA_LNB, I_SA_LNG, I_SA_WK, I_SA_WO, I_SA_WQ, I_SA_WV,
    I_SF_B1, I_SF_B2, I_SF_LNB, I_SF_LNG, I_SF_W1, I_SF_W2,
    I_T_B1, I_T_B2, I_T_W1, I_T_W2, I_TR_B, I_TR_W
};

extern "C" void kernel_launch(void* const* d_in, const int* in_sizes, int n_in,
                              void* d_out, int out_size, void* d_ws, size_t ws_size,
                              hipStream_t stream) {
    (void)in_sizes; (void)n_in; (void)out_size; (void)ws_size;
    const int NLAT = 2048, NTOK = 2304, SEQ = 2306;
    const int M_LAT = 8 * NLAT;     // 16384
    const int M_SEQ = 8 * SEQ;      // 18448
    const int M_TOK = 8 * NTOK;     // 18432

    char* base = (char*)d_ws;
    size_t off = 0;
    auto alloc = [&](size_t bytes) -> char* {
        char* r = base + off;
        off += (bytes + 255) & ~(size_t)255;
        return r;
    };
    auto F = [&](int i) { return (const float*)d_in[i]; };

    // bf16 weight copies, all pre-transposed to [N][K] (K-contiguous)
    u16* w_tr  = (u16*)alloc((size_t)65536 * 2);
    u16* w_cwq = (u16*)alloc((size_t)16384 * 2);
    u16* w_cwk = (u16*)alloc((size_t)16384 * 2);
    u16* w_cwv = (u16*)alloc((size_t)16384 * 2);
    u16* w_cwo = (u16*)alloc((size_t)16384 * 2);
    u16* w_cf1 = (u16*)alloc((size_t)524288 * 2);
    u16* w_cf2 = (u16*)alloc((size_t)262144 * 2);
    u16* w_swq = (u16*)alloc((size_t)6 * 16384 * 2);
    u16* w_swk = (u16*)alloc((size_t)6 * 16384 * 2);
    u16* w_swv = (u16*)alloc((size_t)6 * 16384 * 2);
    u16* w_swo = (u16*)alloc((size_t)6 * 16384 * 2);
    u16* w_sw1 = (u16*)alloc((size_t)6 * 524288 * 2);
    u16* w_sw2 = (u16*)alloc((size_t)6 * 262144 * 2);
    u16* w_dwq = (u16*)alloc((size_t)8192 * 2);
    u16* w_dwk = (u16*)alloc((size_t)8192 * 2);
    u16* w_dwv = (u16*)alloc((size_t)8192 * 2);
    u16* w_dwo = (u16*)alloc((size_t)8192 * 2);
    u16* w_pr  = (u16*)alloc((size_t)524288 * 2);

    float* te      = (float*)alloc((size_t)8 * 256 * 4);
    float* pe      = (float*)alloc((size_t)8 * 256 * 4);
    u16*   trajb   = (u16*)alloc((size_t)M_TOK * 256 * 2);
    float* stacked = (float*)alloc((size_t)M_SEQ * 256 * 4);
    float* lat     = (float*)alloc((size_t)M_LAT * 256 * 4);
    u16*   xa      = (u16*)alloc((size_t)M_SEQ * 256 * 2);
    u16*   xb      = (u16*)alloc((size_t)M_SEQ * 256 * 2);
    u16*   qb      = (u16*)alloc((size_t)M_SEQ * 64 * 2);
    u16*   kb      = (u16*)alloc((size_t)M_SEQ * 64 * 2);
    u16*   vb      = (u16*)alloc((size_t)M_SEQ * 64 * 2);
    u16*   ob      = (u16*)alloc((size_t)M_SEQ * 64 * 2);
    u16*   ub      = (u16*)alloc((size_t)M_LAT * 2048 * 2);
    u16*   hb      = (u16*)alloc((size_t)M_LAT * 1024 * 2);

    auto cvtT = [&](const float* src, u16* dst, int K, int N) {
        cvt_t_kernel<<<(K * N + 255) / 256, 256, 0, stream>>>(src, dst, K, N);
    };
    auto gemm = [&](const u16* A, const u16* W, const float* bias,
                    const float* res, float* oF, u16* oH,
                    int M, int N, int K, int aPhys,
                    int ropb, int ripb, int rskip, int oripb, int oskip) {
        dim3 g((N + 127) / 128, (M + 127) / 128);
        gemm_bf16_kernel<<<g, 256, 0, stream>>>(A, W, bias, res, oF, oH,
                                                M, N, K, aPhys,
                                                ropb, ripb, rskip, oripb, oskip);
    };
    auto gemmS = [&](const u16* A, const u16* W, const float* bias,
                     const float* res, float* oF, u16* oH, int M, int N, int K) {
        gemm(A, W, bias, res, oF, oH, M, N, K, M, M, M, 0, M, 0);
    };
    auto ln = [&](const float* in, u16* out, const float* g, const float* b, int rows) {
        ln_kernel<<<rows, 256, 0, stream>>>(in, out, g, b);
    };

    // ---- convert + transpose weights ----
    cvtT(F(I_TR_W), w_tr, 256, 256);
    cvtT(F(I_CA_WQ), w_cwq, 256, 64);  cvtT(F(I_CA_WK), w_cwk, 256, 64);
    cvtT(F(I_CA_WV), w_cwv, 256, 64);  cvtT(F(I_CA_WO), w_cwo, 64, 256);
    cvtT(F(I_CF_W1), w_cf1, 256, 2048);
    cvtT(F(I_CF_W2), w_cf2, 1024, 256);
    for (int i = 0; i < 6; ++i) {
        cvtT(F(I_SA_WQ) + (size_t)i * 16384, w_swq + (size_t)i * 16384, 256, 64);
        cvtT(F(I_SA_WK) + (size_t)i * 16384, w_swk + (size_t)i * 16384, 256, 64);
        cvtT(F(I_SA_WV) + (size_t)i * 16384, w_swv + (size_t)i * 16384, 256, 64);
        cvtT(F(I_SA_WO) + (size_t)i * 16384, w_swo + (size_t)i * 16384, 64, 256);
        cvtT(F(I_SF_W1) + (size_t)i * 524288, w_sw1 + (size_t)i * 524288, 256, 2048);
        cvtT(F(I_SF_W2) + (size_t)i * 262144, w_sw2 + (size_t)i * 262144, 1024, 256);
    }
    cvtT(F(I_DC_WQ), w_dwq, 256, 32);  cvtT(F(I_DC_WK), w_dwk, 256, 32);
    cvtT(F(I_DC_WV), w_dwv, 256, 32);  cvtT(F(I_DC_WO), w_dwo, 32, 256);
    cvtT(F(I_PR_W), w_pr, 256, 2048);

    // ---- latent init + embeddings ----
    (void)hipMemcpyAsync(lat, d_in[I_XCOND], (size_t)M_LAT * 256 * 4,
                         hipMemcpyDeviceToDevice, stream);
    embed_small_kernel<<<8, 256, 0, stream>>>(
        F(I_TIME), F(I_COND), F(I_T_W1), F(I_T_B1), F(I_T_W2), F(I_T_B2),
        F(I_P_LNG), F(I_P_LNB), F(I_P_W1), F(I_P_B1), F(I_P_W2), F(I_P_B2), te, pe);
    gather_kernel<<<M_TOK, 256, 0, stream>>>((const int*)d_in[I_TRAJ],
                                             F(I_CODEBOOK), trajb);
    // tre -> stacked rows 2..2305 of each batch (out row remap)
    gemm(trajb, w_tr, F(I_TR_B), nullptr, stacked, nullptr,
         M_TOK, 256, 256, M_TOK, NTOK, NTOK, 0, SEQ, 2);
    build_stack_kernel<<<M_SEQ, 256, 0, stream>>>(stacked, te, pe, F(I_POS),
                                                  F(I_ELN_G), F(I_ELN_B));

    // ---- cross attention + FF ----
    ln(lat, xa, F(I_CA_LNG), F(I_CA_LNB), M_LAT);
    ln(stacked, xb, F(I_CA_CLNG), F(I_CA_CLNB), M_SEQ);
    gemmS(xa, w_cwq, nullptr, nullptr, nullptr, qb, M_LAT, 64, 256);
    gemmS(xb, w_cwk, nullptr, nullptr, nullptr, kb, M_SEQ, 64, 256);
    gemmS(xb, w_cwv, nullptr, nullptr, nullptr, vb, M_SEQ, 64, 256);
    attn_kernel<64><<<dim3(16, 8), 256, 0, stream>>>(qb, kb, vb, ob,
                                                     NLAT, SEQ, 0.125f);
    gemmS(ob, w_cwo, F(I_CA_BO), lat, lat, nullptr, M_LAT, 256, 64);

    ln(lat, xa, F(I_CF_LNG), F(I_CF_LNB), M_LAT);
    gemmS(xa, w_cf1, F(I_CF_B1), nullptr, nullptr, ub, M_LAT, 2048, 256);
    geglu_kernel<<<(int)(((size_t)M_LAT * 1024 + 255) / 256), 256, 0, stream>>>(ub, hb, M_LAT);
    gemmS(hb, w_cf2, F(I_CF_B2), lat, lat, nullptr, M_LAT, 256, 1024);

    // ---- self-attention blocks ----
    for (int i = 0; i < 6; ++i) {
        ln(lat, xa, F(I_SA_LNG) + i * 256, F(I_SA_LNB) + i * 256, M_LAT);
        gemmS(xa, w_swq + (size_t)i * 16384, nullptr, nullptr, nullptr, qb, M_LAT, 64, 256);
        gemmS(xa, w_swk + (size_t)i * 16384, nullptr, nullptr, nullptr, kb, M_LAT, 64, 256);
        gemmS(xa, w_swv + (size_t)i * 16384, nullptr, nullptr, nullptr, vb, M_LAT, 64, 256);
        attn_kernel<64><<<dim3(16, 8), 256, 0, stream>>>(qb, kb, vb, ob,
                                                         NLAT, NLAT, 0.125f);
        gemmS(ob, w_swo + (size_t)i * 16384, F(I_SA_BO) + i * 256, lat, lat, nullptr,
              M_LAT, 256, 64);

        ln(lat, xa, F(I_SF_LNG) + i * 256, F(I_SF_LNB) + i * 256, M_LAT);
        gemmS(xa, w_sw1 + (size_t)i * 524288, F(I_SF_B1) + i * 2048, nullptr, nullptr,
              ub, M_LAT, 2048, 256);
        geglu_kernel<<<(int)(((size_t)M_LAT * 1024 + 255) / 256), 256, 0, stream>>>(ub, hb, M_LAT);
        gemmS(hb, w_sw2 + (size_t)i * 262144, F(I_SF_B2) + i * 256, lat, lat, nullptr,
              M_LAT, 256, 1024);
    }

    // ---- decoder cross-attention + output projection ----
    ln(stacked, xa, F(I_DC_LNG), F(I_DC_LNB), M_SEQ);
    ln(lat, xb, F(I_DC_CLNG), F(I_DC_CLNB), M_LAT);
    gemmS(xa, w_dwq, nullptr, nullptr, nullptr, qb, M_SEQ, 32, 256);
    gemmS(xb, w_dwk, nullptr, nullptr, nullptr, kb, M_LAT, 32, 256);
    gemmS(xb, w_dwv, nullptr, nullptr, nullptr, vb, M_LAT, 32, 256);
    attn_kernel<32><<<dim3(19, 8), 256, 0, stream>>>(qb, kb, vb, ob,
                                                     SEQ, NLAT, 0.1767766953f);
    gemmS(ob, w_dwo, F(I_DC_BO), nullptr, nullptr, xb, M_SEQ, 256, 32);
    // pred = dec[:, -2304:] @ pr_w + pr_b (input row remap skips 2 per batch)
    gemm(xb, w_pr, F(I_PR_B), nullptr, (float*)d_out, nullptr,
         M_TOK, 2048, 256, M_SEQ, NTOK, SEQ, 2, NTOK, 0);
}
